// Transformer2_489626272103
// MI455X (gfx1250) — matrix operand order repaired and checked by reading, not compiled, not run on hardware
//
#include <hip/hip_runtime.h>
#include <math.h>

typedef __attribute__((ext_vector_type(16))) __bf16 v16bf;
typedef __attribute__((ext_vector_type(8)))  float  v8f;

#define Hx  1024
#define Sx  4096
#define Bx  8
#define Ex  8
#define Rx  64
#define RHx 256

union BF16Frag {
    v16bf  v;
    __bf16 e[16];
    uint4  q[2];
};

__device__ __forceinline__ float gelu_erf(float x) {
    return 0.5f * x * (1.0f + erff(x * 0.70710678118654752440f));
}

__device__ __forceinline__ v8f zero8() {
    v8f z;
#pragma unroll
    for (int i = 0; i < 8; ++i) z[i] = 0.0f;
    return z;
}

// A-fragment (16x32 bf16, M x K) built from row-major f32 source.
// lane l: m = l&15 ; kb = (l>>4)*8 + 32*kstep
// elements 0..7  = A[m][kb .. kb+7]
// elements 8..15 = A[m][kb+16 .. kb+23]
__device__ __forceinline__ v16bf load_afrag_f32(const float* __restrict__ base,
                                                int ld, int m, int kb) {
    const float4* p0 = (const float4*)(base + (size_t)m * ld + kb);
    const float4* p1 = (const float4*)(base + (size_t)m * ld + kb + 16);
    float4 a = p0[0], b = p0[1], c = p1[0], d = p1[1];
    BF16Frag f;
    f.e[0]  = (__bf16)a.x; f.e[1]  = (__bf16)a.y; f.e[2]  = (__bf16)a.z; f.e[3]  = (__bf16)a.w;
    f.e[4]  = (__bf16)b.x; f.e[5]  = (__bf16)b.y; f.e[6]  = (__bf16)b.z; f.e[7]  = (__bf16)b.w;
    f.e[8]  = (__bf16)c.x; f.e[9]  = (__bf16)c.y; f.e[10] = (__bf16)c.z; f.e[11] = (__bf16)c.w;
    f.e[12] = (__bf16)d.x; f.e[13] = (__bf16)d.y; f.e[14] = (__bf16)d.z; f.e[15] = (__bf16)d.w;
    return f.v;
}

// B-fragment (32x16 bf16, K x N) from n-major (K-contiguous) bf16 source.
// lane l: n = ntile*16 + (l&15); koff includes (l>>4)*8; elements 0..7 = B[koff + j][n], 8..15 = B[koff + 16 + (j-8)][n]
__device__ __forceinline__ v16bf load_bfrag_bf16(const __bf16* __restrict__ base,
                                                 int ldk, int n, int koff) {
    BF16Frag f;
    const uint4* p = (const uint4*)(base + (size_t)n * ldk + koff);
    f.q[0] = p[0];
    f.q[1] = p[2];
    return f.v;
}

// A-fragment from LDS bf16 tile with row length 64 (K dim = R).
__device__ __forceinline__ v16bf load_afrag_lds(const __bf16* gz, int m, int kb) {
    BF16Frag f;
    const uint4* p0 = (const uint4*)(gz + m * Rx + kb);
    const uint4* p1 = (const uint4*)(gz + m * Rx + kb + 16);
    f.q[0] = p0[0];
    f.q[1] = p1[0];
    return f.v;
}

// ---------------------------------------------------------------------------
// Weight prep: f32 -> bf16 with transpose to K-contiguous per output column.
// WdT[e][r][h] = w_down[e][h][r]   (B for down-GEMM: K=H contiguous)
// WuT[e][h][r] = w_up[e][r][h]     (B for up-GEMM:   K=R contiguous)
// ---------------------------------------------------------------------------
__global__ __launch_bounds__(256) void prep_weights(const float* __restrict__ wd,
                                                    const float* __restrict__ wu,
                                                    __bf16* __restrict__ WdT,
                                                    __bf16* __restrict__ WuT) {
    int idx = blockIdx.x * 256 + threadIdx.x;   // grid sized exactly E*H*R/256
    int e   = idx >> 16;                        // / (H*R) = 65536
    int rem = idx & 65535;
    int r  = rem >> 10, h  = rem & 1023;        // WdT[e][r][h]
    int h2 = rem >> 6,  r2 = rem & 63;          // WuT[e][h][r]
    WdT[idx] = (__bf16)wd[((size_t)e << 16) + (size_t)h * Rx + r];
    WuT[idx] = (__bf16)wu[((size_t)e << 16) + (size_t)r2 * Hx + h2];
}

// ---------------------------------------------------------------------------
// Mean pool over S (partial sums + atomics). pooled must be pre-zeroed.
// ---------------------------------------------------------------------------
__global__ __launch_bounds__(256) void pool_kernel(const float* __restrict__ X,
                                                   float* __restrict__ pooled) {
    int h  = blockIdx.y * 256 + threadIdx.x;
    int b  = blockIdx.z;
    int s0 = blockIdx.x * (Sx / 32);
    const float* p = X + ((size_t)b * Sx + s0) * Hx + h;
    float sum = 0.0f;
    for (int i = 0; i < Sx / 32; ++i) sum += p[(size_t)i * Hx];
    atomicAdd(&pooled[b * Hx + h], sum);
}

// ---------------------------------------------------------------------------
// Router: pooled @ r_w1 + b -> LN -> gelu(erf) -> @ r_w2 + b -> softmax -> z
// ---------------------------------------------------------------------------
__global__ __launch_bounds__(256) void router_kernel(const float* __restrict__ pooled_sum,
                                                     const float* __restrict__ r_w1,
                                                     const float* __restrict__ r_b1,
                                                     const float* __restrict__ ln_g,
                                                     const float* __restrict__ ln_b,
                                                     const float* __restrict__ r_w2,
                                                     const float* __restrict__ r_b2,
                                                     float* __restrict__ zbuf,
                                                     float* __restrict__ zout) {
    __shared__ float red[RHx];
    __shared__ float hbuf[Bx][RHx];
    __shared__ float lg[Bx * Ex];
    const int t = threadIdx.x;
    const float invS = 1.0f / (float)Sx;

    for (int b = 0; b < Bx; ++b) {
        float acc = r_b1[t];
        for (int h = 0; h < Hx; ++h)
            acc += pooled_sum[b * Hx + h] * invS * r_w1[h * RHx + t];
        red[t] = acc; __syncthreads();
        for (int off = 128; off > 0; off >>= 1) {
            if (t < off) red[t] += red[t + off];
            __syncthreads();
        }
        float mu = red[0] * (1.0f / RHx); __syncthreads();
        float d = acc - mu;
        red[t] = d * d; __syncthreads();
        for (int off = 128; off > 0; off >>= 1) {
            if (t < off) red[t] += red[t + off];
            __syncthreads();
        }
        float var = red[0] * (1.0f / RHx); __syncthreads();
        float hn = d * rsqrtf(var + 1e-5f) * ln_g[t] + ln_b[t];
        hbuf[b][t] = gelu_erf(hn);
    }
    __syncthreads();
    if (t < Bx * Ex) {
        int b = t / Ex, e = t % Ex;
        float acc = r_b2[e];
        for (int rh = 0; rh < RHx; ++rh) acc += hbuf[b][rh] * r_w2[rh * Ex + e];
        lg[t] = acc;
    }
    __syncthreads();
    if (t < Bx) {
        int b = t;
        float mx = -1e30f;
        for (int e = 0; e < Ex; ++e) mx = fmaxf(mx, lg[b * Ex + e]);
        float s = 0.0f, ex[Ex];
        for (int e = 0; e < Ex; ++e) { ex[e] = expf(lg[b * Ex + e] - mx); s += ex[e]; }
        float inv = 1.0f / s;
        for (int e = 0; e < Ex; ++e) {
            float v = ex[e] * inv;
            zbuf[b * Ex + e] = v;
            zout[b * Ex + e] = v;
        }
    }
}

// ---------------------------------------------------------------------------
// Fused dense-expert pass. One wave per 16-row tile of [B*S, H].
// Phase 1: down[e] = gelu(X @ WdT[e]^n-major) * z[b,e]  -> LDS (bf16)
// Phase 2: out = X + sum_e (z*g)[e] @ WuT[e]^n-major
// ---------------------------------------------------------------------------
__global__ __launch_bounds__(32) void expert_kernel(const float*  __restrict__ X,
                                                    const __bf16* __restrict__ WdT,
                                                    const __bf16* __restrict__ WuT,
                                                    const float*  __restrict__ z,
                                                    float* __restrict__ out) {
    __shared__ __bf16 gz[Ex][16][Rx];          // 16 KB: z-scaled gelu(down)

    const int lane = threadIdx.x;
    const int tile = blockIdx.x;               // 0..2047
    const int r0   = tile * 16;                // global row (b*S + s)
    const int b    = r0 / Sx;                  // tiles never straddle batches
    const float* Xt = X + (size_t)r0 * Hx;

    const int mA   = lane & 15;
    const int half = lane >> 4;

    // ---------------- Phase 1: down-projection, 2 groups of 4 experts -------
    for (int eg = 0; eg < 2; ++eg) {
        v8f acc[4][4];
#pragma unroll
        for (int ei = 0; ei < 4; ++ei)
#pragma unroll
            for (int nt = 0; nt < 4; ++nt) acc[ei][nt] = zero8();

        for (int ks = 0; ks < Hx / 32; ++ks) {
            v16bf A = load_afrag_f32(Xt, Hx, mA, half * 8 + ks * 32);
#pragma unroll
            for (int ei = 0; ei < 4; ++ei) {
                const __bf16* Wd = WdT + (size_t)(eg * 4 + ei) * Rx * Hx;
#pragma unroll
                for (int nt = 0; nt < 4; ++nt) {
                    v16bf Bf = load_bfrag_bf16(Wd, Hx, nt * 16 + mA, ks * 32 + half * 8);
                    acc[ei][nt] = __builtin_amdgcn_wmma_f32_16x16x32_bf16(
                        false, A, false, Bf, (short)0, acc[ei][nt], false, false);
                }
            }
        }
        // gelu + z-scale, write to LDS (C/D layout: lane holds col n, rows half*8+j)
#pragma unroll
        for (int ei = 0; ei < 4; ++ei) {
            const int e = eg * 4 + ei;
            const float ze = z[b * Ex + e];
#pragma unroll
            for (int nt = 0; nt < 4; ++nt) {
                const int n = nt * 16 + mA;
#pragma unroll
                for (int j = 0; j < 8; ++j)
                    gz[e][half * 8 + j][n] = (__bf16)(ze * gelu_erf(acc[ei][nt][j]));
            }
        }
    }
    __syncthreads();   // single wave: ensures DScnt drained before cross-lane reads

    // ---------------- Phase 2: up-projection + weighted sum + residual ------
    for (int ht = 0; ht < Hx / 16; ++ht) {
        v8f acc = zero8();
#pragma unroll
        for (int e = 0; e < Ex; ++e) {
            const __bf16* Wu = WuT + (size_t)e * Hx * Rx;
#pragma unroll
            for (int ks = 0; ks < 2; ++ks) {
                v16bf A  = load_afrag_lds(&gz[e][0][0], mA, half * 8 + ks * 32);
                v16bf Bf = load_bfrag_bf16(Wu, Rx, ht * 16 + mA, ks * 32 + half * 8);
                acc = __builtin_amdgcn_wmma_f32_16x16x32_bf16(
                    false, A, false, Bf, (short)0, acc, false, false);
            }
        }
        const int n = ht * 16 + mA;
#pragma unroll
        for (int j = 0; j < 8; ++j) {
            const size_t idx = (size_t)(r0 + half * 8 + j) * Hx + n;
            out[idx] = acc[j] + X[idx];
        }
    }
}

// ---------------------------------------------------------------------------
extern "C" void kernel_launch(void* const* d_in, const int* in_sizes, int n_in,
                              void* d_out, int out_size, void* d_ws, size_t ws_size,
                              hipStream_t stream) {
    const float* X   = (const float*)d_in[0];
    const float* wd  = (const float*)d_in[1];
    const float* wu  = (const float*)d_in[2];
    const float* rw1 = (const float*)d_in[3];
    const float* rb1 = (const float*)d_in[4];
    const float* lng = (const float*)d_in[5];
    const float* lnb = (const float*)d_in[6];
    const float* rw2 = (const float*)d_in[7];
    const float* rb2 = (const float*)d_in[8];
    float* out = (float*)d_out;

    char* ws = (char*)d_ws;
    float*  pooled = (float*)ws;                               // 32 KB
    float*  zbuf   = (float*)(ws + 32768);                     // 256 B
    __bf16* WdT    = (__bf16*)(ws + 65536);                    // 1 MB
    __bf16* WuT    = (__bf16*)(ws + 65536 + (size_t)Ex * Hx * Rx * 2);

    hipMemsetAsync(pooled, 0, (size_t)Bx * Hx * sizeof(float), stream);

    prep_weights<<<(Ex * Hx * Rx) / 256, 256, 0, stream>>>(wd, wu, WdT, WuT);
    pool_kernel<<<dim3(32, Hx / 256, Bx), 256, 0, stream>>>(X, pooled);
    router_kernel<<<1, 256, 0, stream>>>(pooled, rw1, rb1, lng, lnb, rw2, rb2,
                                         zbuf, out + (size_t)Bx * Sx * Hx);
    expert_kernel<<<(Bx * Sx) / 16, 32, 0, stream>>>(X, WdT, WuT, zbuf, out);
}
